// GFlowNet_89094801588524
// MI455X (gfx1250) — matrix operand order repaired and checked by reading, not compiled, not run on hardware
//
#include <hip/hip_runtime.h>
#include <hip/hip_bf16.h>

// ---------------- constants for this workload ----------------
#define S_STEPS 96
#define DMODEL  512
#define NHEAD   8
#define DHEAD   64
#define LAYERS  6
#define FDIM    2048
#define VOCAB   32000
#define NTOK    (S_STEPS * S_STEPS)   // 9216 rows in the activation tensor
#define MPAD    128                   // padded row count for the logits GEMM

// ---------------- WMMA types ----------------
typedef __attribute__((ext_vector_type(16))) __bf16 v16bf;
typedef __attribute__((ext_vector_type(8)))  float  v8f;

union FragAB {
    v16bf v;
    unsigned int u[8];
};

// round-to-nearest-even f32 -> bf16 bits
__device__ __forceinline__ unsigned int f2bf(float f) {
    unsigned int u = __float_as_uint(f);
    u += 0x7FFFu + ((u >> 16) & 1u);
    return u >> 16;
}
__device__ __forceinline__ unsigned int pack_bf16(float lo, float hi) {
    return f2bf(lo) | (f2bf(hi) << 16);
}

#if __has_builtin(__builtin_amdgcn_s_wait_asynccnt)
#define WAIT_ASYNCCNT(n) __builtin_amdgcn_s_wait_asynccnt(n)
#else
#define WAIT_ASYNCCNT(n) asm volatile("s_wait_asynccnt %0" :: "n"(n) : "memory")
#endif

// async global -> LDS copy of 16 bytes (ASYNCcnt-tracked, no VGPR data path)
__device__ __forceinline__ void async_copy_b128(unsigned lds_addr,
                                                unsigned long long gaddr) {
    asm volatile("global_load_async_to_lds_b128 %0, %1, off"
                 :: "v"(lds_addr), "v"(gaddr) : "memory");
}

// ---------------- f32 -> packed bf16x2 bulk conversion (weights) ------------
__global__ void cvt_bf16_kernel(const float* __restrict__ in,
                                unsigned int* __restrict__ out, long n2) {
    long i = (long)blockIdx.x * blockDim.x + threadIdx.x;
    if (i >= n2) return;
    float2 v = *(const float2*)(in + 2 * i);
    out[i] = pack_bf16(v.x, v.y);
}

// ---------------- embedding + pos0 fill: f32 x and packed-bf16 mirror -------
__global__ void embed_kernel(const int* __restrict__ tokens,
                             const float* __restrict__ emb,
                             const float* __restrict__ pos0,
                             float* __restrict__ x,
                             unsigned int* __restrict__ xbf) {
    long idx = (long)blockIdx.x * blockDim.x + threadIdx.x;   // pair index
    if (idx >= (long)NTOK * (DMODEL / 2)) return;
    int dp  = (int)(idx & (DMODEL / 2 - 1));
    int row = (int)(idx >> 8);
    int p   = row % S_STEPS;
    int tok = tokens[p];
    float2 e = *(const float2*)(emb + (long)tok * DMODEL + 2 * dp);
    float2 q = *(const float2*)(pos0 + 2 * dp);
    float2 v; v.x = e.x + q.x; v.y = e.y + q.y;
    *(float2*)(x + (long)row * DMODEL + 2 * dp) = v;
    xbf[idx] = pack_bf16(v.x, v.y);
}

// ---------------- bf16 WMMA GEMM with async-LDS double buffering ------------
// C[m][n] = sum_k A[m][k] * W[n][k] (+ bias[n]) (opt. ReLU, opt. bf16 store)
// A: packed bf16 pairs [M, K/2] dwords; W: bf16 bits [N,K] row-major.
// Requirements: M % 64 == 0, N % 128 == 0, K % 32 == 0 (caller pads).
// Block: 128 threads = 4 waves in 2x2; block tile 64x128, each wave 32x64.
// Tiles are staged with GLOBAL_LOAD_ASYNC_TO_LDS_B128 (ASYNCcnt), so no
// VGPRs are burned on staging and chunk i+1 streams in during the WMMAs.
#define BM 64
#define BN 128
#define BK 32
#define LDA 20   // dword stride per A row (16 data + 4 pad; keeps 16B alignment)
#define LDB 20   // dword stride per B row

__global__ __launch_bounds__(128)
void gemm_bf16_kernel(const unsigned int* __restrict__ Apk,
                      const unsigned short* __restrict__ W,
                      void* __restrict__ Cout,
                      const float* __restrict__ bias,
                      int M, int N, int K, int relu, int storeBf16) {
    __shared__ __align__(16) unsigned int As32[2][BM * LDA];
    __shared__ __align__(16) unsigned int Bs32[2][BN * LDB];

    const int tid   = threadIdx.x;
    const int wave  = tid >> 5;
    const int lane  = tid & 31;
    const int waveM = wave >> 1;   // 0..1
    const int waveN = wave & 1;    // 0..1
    const long row0 = (long)blockIdx.y * BM;
    const long col0 = (long)blockIdx.x * BN;
    const long Kd   = K >> 1;      // A row stride in dwords

    v8f acc[2][4];
    #pragma unroll
    for (int i = 0; i < 2; ++i)
        #pragma unroll
        for (int j = 0; j < 4; ++j)
            #pragma unroll
            for (int e = 0; e < 8; ++e) acc[i][j][e] = 0.0f;

    // per-thread async staging slots (both A and B advance 64 B per K-chunk)
    unsigned            ldsA[2], ldsB[4];
    unsigned long long  gA[2],  gB[4];
    #pragma unroll
    for (int j = 0; j < 2; ++j) {                   // 64x16dw A tile
        int e = j * 512 + tid * 4;
        int r = e >> 4, c = e & 15;                 // c multiple of 4
        ldsA[j] = (unsigned)(size_t)&As32[0][r * LDA + c];
        gA[j]   = (unsigned long long)(size_t)(Apk + (row0 + r) * Kd + c);
    }
    #pragma unroll
    for (int j = 0; j < 4; ++j) {                   // 128x16dw B tile
        int e = j * 512 + tid * 4;
        int n = e >> 4, kp = e & 15;                // kp multiple of 4 (dwords)
        ldsB[j] = (unsigned)(size_t)&Bs32[0][n * LDB + kp];
        gB[j]   = (unsigned long long)(size_t)(W + (col0 + n) * K + kp * 2);
    }
    const unsigned aBuf = sizeof(unsigned int) * BM * LDA;   // 5120 B
    const unsigned bBuf = sizeof(unsigned int) * BN * LDB;   // 10240 B

    auto issue_async = [&](int it) {
        unsigned long long koff = (unsigned long long)it * 64u;  // bytes
        unsigned ab = (it & 1) ? aBuf : 0u;
        unsigned bb = (it & 1) ? bBuf : 0u;
        #pragma unroll
        for (int j = 0; j < 2; ++j) async_copy_b128(ldsA[j] + ab, gA[j] + koff);
        #pragma unroll
        for (int j = 0; j < 4; ++j) async_copy_b128(ldsB[j] + bb, gB[j] + koff);
    };

    const int nIt = K / BK;
    const int mr = lane & 15;   // A fragment row
    const int kh = lane >> 4;   // K-half selector
    const int nc = lane & 15;   // B/C column within 16-tile

    issue_async(0);

    #pragma unroll 2
    for (int it = 0; it < nIt; ++it) {
        if (it + 1 < nIt) {
            issue_async(it + 1);        // streams during this chunk's WMMAs
            WAIT_ASYNCCNT(6);           // in-order: chunk `it` is resident
        } else {
            WAIT_ASYNCCNT(0);
        }
        __syncthreads();                // all waves' tiles visible
        const int buf = it & 1;

        FragAB af[2];
        #pragma unroll
        for (int im = 0; im < 2; ++im) {
            int m = waveM * 32 + im * 16 + mr;
            uint4 lo = *(const uint4*)&As32[buf][m * LDA + kh * 4];
            uint4 hi = *(const uint4*)&As32[buf][m * LDA + 8 + kh * 4];
            af[im].u[0] = lo.x; af[im].u[1] = lo.y; af[im].u[2] = lo.z; af[im].u[3] = lo.w;
            af[im].u[4] = hi.x; af[im].u[5] = hi.y; af[im].u[6] = hi.z; af[im].u[7] = hi.w;
        }
        #pragma unroll
        for (int jn = 0; jn < 4; ++jn) {
            int n = waveN * 64 + jn * 16 + nc;
            FragAB bfrag;
            uint4 lo = *(const uint4*)&Bs32[buf][n * LDB + kh * 4];
            uint4 hi = *(const uint4*)&Bs32[buf][n * LDB + 8 + kh * 4];
            bfrag.u[0] = lo.x; bfrag.u[1] = lo.y; bfrag.u[2] = lo.z; bfrag.u[3] = lo.w;
            bfrag.u[4] = hi.x; bfrag.u[5] = hi.y; bfrag.u[6] = hi.z; bfrag.u[7] = hi.w;
            #pragma unroll
            for (int im = 0; im < 2; ++im)
                acc[im][jn] = __builtin_amdgcn_wmma_f32_16x16x32_bf16(
                    false, af[im].v, false, bfrag.v,
                    (short)0, acc[im][jn], false, false);
        }
        __syncthreads();                // done reading buf before it's refilled
    }

    // epilogue: C f32 16x16 layout — VGPR j holds M=j (lanes 0-15) / M=j+8 (16-31)
    const int mh = lane >> 4;
    #pragma unroll
    for (int im = 0; im < 2; ++im) {
        #pragma unroll
        for (int jn = 0; jn < 4; ++jn) {
            long cgl = col0 + waveN * 64 + jn * 16 + nc;
            float bb = bias ? bias[cgl] : 0.0f;
            long rbase = row0 + waveM * 32 + im * 16 + 8 * mh;
            if (storeBf16) {
                unsigned short* p = (unsigned short*)Cout + rbase * N + cgl;
                #pragma unroll
                for (int j = 0; j < 8; ++j) {
                    float v = acc[im][jn][j] + bb;
                    if (relu) v = fmaxf(v, 0.0f);
                    p[(long)j * N] = (unsigned short)f2bf(v);
                }
            } else {
                float* p = (float*)Cout + rbase * N + cgl;
                #pragma unroll
                for (int j = 0; j < 8; ++j) {
                    float v = acc[im][jn][j] + bb;
                    if (relu) v = fmaxf(v, 0.0f);
                    p[(long)j * N] = v;
                }
            }
        }
    }
}

// ---------------- masked self-attention (bf16 output for O-proj GEMM) -------
__global__ __launch_bounds__(128)
void attn_kernel(const float* __restrict__ qkv, unsigned short* __restrict__ outp) {
    __shared__ float Ks[S_STEPS][DHEAD + 1];
    __shared__ float Vs[S_STEPS][DHEAD + 1];
    __shared__ float qs[DHEAD];
    __shared__ float sc[S_STEPS];
    __shared__ float red[128];

    const int s   = blockIdx.x / NHEAD;
    const int h   = blockIdx.x % NHEAD;
    const int tid = threadIdx.x;
    const int nk  = s + 1;
    const float scale = 0.125f; // 1/sqrt(64)

    for (int e = tid; e < S_STEPS * DHEAD; e += 128) {
        int p = e >> 6, d = e & 63;
        long base = (long)(s * S_STEPS + p) * (3 * DMODEL) + h * DHEAD + d;
        Ks[p][d] = qkv[base + DMODEL];
        Vs[p][d] = qkv[base + 2 * DMODEL];
    }
    __syncthreads();

    for (int q = 0; q < S_STEPS; ++q) {
        if (tid < DHEAD)
            qs[tid] = qkv[(long)(s * S_STEPS + q) * (3 * DMODEL) + h * DHEAD + tid];
        __syncthreads();

        float v = -1e30f;
        if (tid < nk) {
            float d0 = 0.0f;
            #pragma unroll 8
            for (int d = 0; d < DHEAD; ++d) d0 += qs[d] * Ks[tid][d];
            v = d0 * scale;
            sc[tid] = v;
        }
        red[tid] = v;
        __syncthreads();
        for (int st = 64; st > 0; st >>= 1) {
            if (tid < st) red[tid] = fmaxf(red[tid], red[tid + st]);
            __syncthreads();
        }
        float mx = red[0];
        __syncthreads();

        float ex = 0.0f;
        if (tid < nk) { ex = __expf(sc[tid] - mx); sc[tid] = ex; }
        red[tid] = ex;
        __syncthreads();
        for (int st = 64; st > 0; st >>= 1) {
            if (tid < st) red[tid] += red[tid + st];
            __syncthreads();
        }
        float inv = 1.0f / red[0];
        __syncthreads();

        if (tid < DHEAD) {
            float o = 0.0f;
            for (int k = 0; k < nk; ++k) o += sc[k] * Vs[k][tid];
            outp[(long)(s * S_STEPS + q) * DMODEL + h * DHEAD + tid] =
                (unsigned short)f2bf(o * inv);
        }
        __syncthreads();
    }
}

// ---------------- layernorm: x = LN(x + add); also writes packed bf16 -------
__global__ __launch_bounds__(128)
void ln_kernel(float* __restrict__ x, const float* __restrict__ add,
               int broadcastAdd, const float* __restrict__ g,
               const float* __restrict__ b, unsigned int* __restrict__ xbf) {
    __shared__ float red[128];
    const int row = blockIdx.x;
    const int tid = threadIdx.x;
    const int d0  = tid * 4;
    float* xp = x + (long)row * DMODEL;
    const float* ap = broadcastAdd ? add : add + (long)row * DMODEL;

    float4 xv = *(const float4*)(xp + d0);
    float4 av = *(const float4*)(ap + d0);
    float y[4] = { xv.x + av.x, xv.y + av.y, xv.z + av.z, xv.w + av.w };

    float sum = y[0] + y[1] + y[2] + y[3];
    red[tid] = sum;
    __syncthreads();
    for (int st = 64; st > 0; st >>= 1) {
        if (tid < st) red[tid] += red[tid + st];
        __syncthreads();
    }
    float mean = red[0] * (1.0f / DMODEL);
    __syncthreads();

    float vs = 0.0f;
    #pragma unroll
    for (int i = 0; i < 4; ++i) { float t = y[i] - mean; vs += t * t; }
    red[tid] = vs;
    __syncthreads();
    for (int st = 64; st > 0; st >>= 1) {
        if (tid < st) red[tid] += red[tid + st];
        __syncthreads();
    }
    float r = rsqrtf(red[0] * (1.0f / DMODEL) + 1e-5f);

    float4 gv = *(const float4*)(g + d0);
    float4 bv = *(const float4*)(b + d0);
    float4 o;
    o.x = (y[0] - mean) * r * gv.x + bv.x;
    o.y = (y[1] - mean) * r * gv.y + bv.y;
    o.z = (y[2] - mean) * r * gv.z + bv.z;
    o.w = (y[3] - mean) * r * gv.w + bv.w;
    *(float4*)(xp + d0) = o;
    uint2 p;
    p.x = pack_bf16(o.x, o.y);
    p.y = pack_bf16(o.z, o.w);
    *(uint2*)(xbf + (long)row * (DMODEL / 2) + tid * 2) = p;
}

// ---------------- cross-attention constant vector ----------------
__global__ void term_kernel(const float* __restrict__ ca_qkv_w_l,
                            const float* __restrict__ ca_qkv_b_l,
                            const float* __restrict__ mem,
                            float* __restrict__ term) {
    int j = blockIdx.x * blockDim.x + threadIdx.x;
    if (j >= DMODEL) return;
    float acc = ca_qkv_b_l[2 * DMODEL + j];
    const float* wrow = ca_qkv_w_l + (long)(2 * DMODEL + j) * DMODEL;
    for (int d = 0; d < DMODEL; ++d) acc += wrow[d] * mem[d];
    term[j] = acc;
}

__global__ void cross_kernel(const float* __restrict__ term,
                             const float* __restrict__ ca_o_w_l,
                             const float* __restrict__ ca_o_b_l,
                             float* __restrict__ cross) {
    int i = blockIdx.x * blockDim.x + threadIdx.x;
    if (i >= DMODEL) return;
    float acc = ca_o_b_l[i];
    const float* wrow = ca_o_w_l + (long)i * DMODEL;
    for (int j = 0; j < DMODEL; ++j) acc += wrow[j] * term[j];
    cross[i] = acc;
}

// ------- gather last = x[s, s, :] as packed bf16, zero-padded to MPAD -------
__global__ void gather_kernel(const float* __restrict__ x,
                              unsigned int* __restrict__ lastbf) {
    int idx = blockIdx.x * blockDim.x + threadIdx.x;   // pair index
    if (idx >= MPAD * (DMODEL / 2)) return;
    int s = idx >> 8, dp = idx & (DMODEL / 2 - 1);
    float2 v = {0.0f, 0.0f};
    if (s < S_STEPS)
        v = *(const float2*)(x + (long)(s * S_STEPS + s) * DMODEL + 2 * dp);
    lastbf[idx] = pack_bf16(v.x, v.y);
}

// ---------------- log-softmax gather ----------------
__global__ __launch_bounds__(256)
void logprob_kernel(const float* __restrict__ logits,
                    const int* __restrict__ tokens, float* __restrict__ out) {
    __shared__ float red[256];
    const int s   = blockIdx.x;
    const int tid = threadIdx.x;
    const float* lp = logits + (long)s * VOCAB;

    float mx = -1e30f;
    for (int j = tid; j < VOCAB; j += 256) mx = fmaxf(mx, lp[j]);
    red[tid] = mx;
    __syncthreads();
    for (int st = 128; st > 0; st >>= 1) {
        if (tid < st) red[tid] = fmaxf(red[tid], red[tid + st]);
        __syncthreads();
    }
    mx = red[0];
    __syncthreads();

    float sum = 0.0f;
    for (int j = tid; j < VOCAB; j += 256) sum += __expf(lp[j] - mx);
    red[tid] = sum;
    __syncthreads();
    for (int st = 128; st > 0; st >>= 1) {
        if (tid < st) red[tid] += red[tid + st];
        __syncthreads();
    }
    if (tid == 0) {
        int tok = tokens[s + 1];
        out[s] = lp[tok] - mx - logf(red[0]);
    }
}

// ---------------- host side ----------------
static inline void launch_gemm(const unsigned int* Apk, const unsigned short* W,
                               void* C, const float* bias, int M, int N, int K,
                               int relu, int storeBf16, hipStream_t stream) {
    dim3 grid(N / BN, M / BM);
    gemm_bf16_kernel<<<grid, 128, 0, stream>>>(Apk, W, C, bias, M, N, K, relu, storeBf16);
}

extern "C" void kernel_launch(void* const* d_in, const int* in_sizes, int n_in,
                              void* d_out, int out_size, void* d_ws, size_t ws_size,
                              hipStream_t stream) {
    (void)in_sizes; (void)n_in; (void)out_size; (void)ws_size;

    const int*   tokens   = (const int*)d_in[0];
    const float* emb      = (const float*)d_in[1];
    const float* pos0     = (const float*)d_in[2];
    const float* mem      = (const float*)d_in[3];
    const float* sa_qkv_w = (const float*)d_in[4];
    const float* sa_qkv_b = (const float*)d_in[5];
    const float* sa_o_w   = (const float*)d_in[6];
    const float* sa_o_b   = (const float*)d_in[7];
    const float* ca_qkv_w = (const float*)d_in[8];
    const float* ca_qkv_b = (const float*)d_in[9];
    const float* ca_o_w   = (const float*)d_in[10];
    const float* ca_o_b   = (const float*)d_in[11];
    const float* ln_w     = (const float*)d_in[12];
    const float* ln_b     = (const float*)d_in[13];
    const float* ff_w1    = (const float*)d_in[14];
    const float* ff_b1    = (const float*)d_in[15];
    const float* ff_w2    = (const float*)d_in[16];
    const float* ff_b2    = (const float*)d_in[17];
    const float* out_w    = (const float*)d_in[18];
    const float* out_b    = (const float*)d_in[19];
    float* outp = (float*)d_out;

    // workspace layout
    char* ws = (char*)d_ws;
    size_t off = 0;
    auto alloc = [&](size_t bytes) -> void* {
        void* p = ws + off;
        off = (off + bytes + 255) & ~(size_t)255;
        return p;
    };
    const size_t nQKV = (size_t)LAYERS * 3 * DMODEL * DMODEL;
    const size_t nO   = (size_t)LAYERS * DMODEL * DMODEL;
    const size_t nW1  = (size_t)LAYERS * FDIM * DMODEL;
    const size_t nW2  = (size_t)LAYERS * DMODEL * FDIM;
    const size_t nWo  = (size_t)VOCAB * DMODEL;

    unsigned short* Wqkv = (unsigned short*)alloc(nQKV * 2);
    unsigned short* Wo   = (unsigned short*)alloc(nO * 2);
    unsigned short* W1   = (unsigned short*)alloc(nW1 * 2);
    unsigned short* W2   = (unsigned short*)alloc(nW2 * 2);
    unsigned short* Wout = (unsigned short*)alloc(nWo * 2);
    float*          x        = (float*)alloc((size_t)NTOK * DMODEL * 4);
    unsigned int*   xbf      = (unsigned int*)alloc((size_t)NTOK * DMODEL / 2 * 4);
    float*          qkv_f32  = (float*)alloc((size_t)NTOK * 3 * DMODEL * 4);
    unsigned short* attn_bf  = (unsigned short*)alloc((size_t)NTOK * DMODEL * 2);
    float*          proj_f32 = (float*)alloc((size_t)NTOK * DMODEL * 4);  // O-proj / FF2 out
    unsigned short* hid_bf   = (unsigned short*)alloc((size_t)NTOK * FDIM * 2);
    float*          term     = (float*)alloc(DMODEL * 4);
    float*          crossv   = (float*)alloc(DMODEL * 4);
    unsigned int*   lastbf   = (unsigned int*)alloc((size_t)MPAD * DMODEL / 2 * 4);
    float*          logits   = (float*)alloc((size_t)MPAD * VOCAB * 4);

    // 1) weight conversions f32 -> packed bf16
    auto cvt = [&](const float* src, unsigned short* dst, size_t n) {
        long n2 = (long)(n / 2);
        cvt_bf16_kernel<<<(unsigned)((n2 + 255) / 256), 256, 0, stream>>>(
            src, (unsigned int*)dst, n2);
    };
    cvt(sa_qkv_w, Wqkv, nQKV);
    cvt(sa_o_w,   Wo,   nO);
    cvt(ff_w1,    W1,   nW1);
    cvt(ff_w2,    W2,   nW2);
    cvt(out_w,    Wout, nWo);

    // 2) embeddings (f32 + bf16 mirror)
    {
        long n2 = (long)NTOK * DMODEL / 2;
        embed_kernel<<<(unsigned)((n2 + 255) / 256), 256, 0, stream>>>(
            tokens, emb, pos0, x, xbf);
    }

    // 3) layers
    for (int l = 0; l < LAYERS; ++l) {
        const unsigned short* wqkv_l = Wqkv + (size_t)l * 3 * DMODEL * DMODEL;
        const unsigned short* wo_l   = Wo   + (size_t)l * DMODEL * DMODEL;
        const unsigned short* w1_l   = W1   + (size_t)l * FDIM * DMODEL;
        const unsigned short* w2_l   = W2   + (size_t)l * DMODEL * FDIM;
        const float* qkvb_l = sa_qkv_b + (size_t)l * 3 * DMODEL;
        const float* ob_l   = sa_o_b   + (size_t)l * DMODEL;
        const float* f1b_l  = ff_b1    + (size_t)l * FDIM;
        const float* f2b_l  = ff_b2    + (size_t)l * DMODEL;

        // QKV projection: [9216,512] x [512,1536] -> f32 (softmax wants f32)
        launch_gemm(xbf, wqkv_l, qkv_f32, qkvb_l, NTOK, 3 * DMODEL, DMODEL, 0, 0, stream);
        // masked attention -> bf16 (GEMM-A only)
        attn_kernel<<<S_STEPS * NHEAD, 128, 0, stream>>>(qkv_f32, attn_bf);
        // O projection: [9216,512] x [512,512] -> f32 residual
        launch_gemm((const unsigned int*)attn_bf, wo_l, proj_f32, ob_l,
                    NTOK, DMODEL, DMODEL, 0, 0, stream);
        // x = LN(x + attn_out), refresh bf16 mirror
        ln_kernel<<<NTOK, 128, 0, stream>>>(x, proj_f32, 0,
            ln_w + ((size_t)l * 3 + 0) * DMODEL, ln_b + ((size_t)l * 3 + 0) * DMODEL, xbf);
        // cross-attn constant vector
        term_kernel<<<2, 256, 0, stream>>>(ca_qkv_w + (size_t)l * 3 * DMODEL * DMODEL,
                                           ca_qkv_b + (size_t)l * 3 * DMODEL, mem, term);
        cross_kernel<<<2, 256, 0, stream>>>(term, ca_o_w + (size_t)l * DMODEL * DMODEL,
                                            ca_o_b + (size_t)l * DMODEL, crossv);
        // x = LN(x + cross)
        ln_kernel<<<NTOK, 128, 0, stream>>>(x, crossv, 1,
            ln_w + ((size_t)l * 3 + 1) * DMODEL, ln_b + ((size_t)l * 3 + 1) * DMODEL, xbf);
        // FFN up + ReLU: [9216,512] x [512,2048] -> bf16 (GEMM-A only)
        launch_gemm(xbf, w1_l, hid_bf, f1b_l, NTOK, FDIM, DMODEL, 1, 1, stream);
        // FFN down: [9216,2048] x [2048,512] -> f32 residual
        launch_gemm((const unsigned int*)hid_bf, w2_l, proj_f32, f2b_l,
                    NTOK, DMODEL, FDIM, 0, 0, stream);
        // x = LN(x + ffn_out)
        ln_kernel<<<NTOK, 128, 0, stream>>>(x, proj_f32, 0,
            ln_w + ((size_t)l * 3 + 2) * DMODEL, ln_b + ((size_t)l * 3 + 2) * DMODEL, xbf);
    }

    // 4) gather diagonal rows as packed bf16 (zero-padded to 128 rows)
    gather_kernel<<<(MPAD * DMODEL / 2 + 255) / 256, 256, 0, stream>>>(x, lastbf);
    // 5) logits: [128,512] x [512,32000]
    launch_gemm(lastbf, Wout, logits, out_b, MPAD, VOCAB, DMODEL, 0, 0, stream);
    // 6) log-softmax + token gather (reads only the 96 real rows)
    logprob_kernel<<<S_STEPS, 256, 0, stream>>>(logits, tokens, outp);
}
